// CopyModel_39135742001212
// MI455X (gfx1250) — compile-verified, matrix operand order checked by reading, not run on hardware
//
#include <hip/hip_runtime.h>
#include <hip/hip_bf16.h>

// ---------- CDNA5 WMMA types ----------
typedef __attribute__((ext_vector_type(16))) _Float16 v16h;
typedef __attribute__((ext_vector_type(8)))  _Float16 v8h;
typedef __attribute__((ext_vector_type(8)))  float    v8f;

// Model dims
#define BATCH 2
#define SEQ 2048
#define ROWS 4096           // BATCH*SEQ
#define HID 1024
#define STATE 4096
#define PROJ 16384          // 4*STATE
#define VOCAB 62
#define NCHUNK 32
#define CHUNKT 64           // NCHUNK*CHUNKT == SEQ

// LDS tile row stride: 64B payload (32 halves) + 16B pad -> conflict-free ds_load_b128
#define LROW 80

__device__ __forceinline__ v8f wmma32(v16h a, v16h b, v8f c) {
  return __builtin_amdgcn_wmma_f32_16x16x32_f16(
      /*neg_a=*/false, a, /*neg_b=*/false, b,
      /*c_mod=*/(short)0, c, /*reuse_a=*/false, /*reuse_b=*/false);
}

// gfx1250 async global->LDS copy, 16B per lane, tracked by ASYNCcnt.
__device__ __forceinline__ void async_copy16(unsigned lds_addr, const void* gaddr) {
  asm volatile("global_load_async_to_lds_b128 %0, %1, off"
               :: "v"(lds_addr), "v"(gaddr) : "memory");
}
__device__ __forceinline__ void wait_async0() {
  asm volatile("s_wait_asynccnt 0" ::: "memory");
}

// Load a 16x32 f16 WMMA fragment from a row-major global matrix (ld halves/row).
// ISA layout: lanes 0-15: rows 0-15, K0..7 (VGPR0-3) + K16..23 (VGPR4-7);
// lanes 16-31: same rows, K8..15 + K24..31. Two contiguous 16B loads per lane.
__device__ __forceinline__ v16h load_frag(const _Float16* base, int ld) {
  int lane = threadIdx.x & 31;
  int r = lane & 15;
  int hi = lane >> 4;
  const _Float16* p = base + (size_t)r * ld + hi * 8;
  union { v16h v; v8h h[2]; } u;
  u.h[0] = *(const v8h*)(p);
  u.h[1] = *(const v8h*)(p + 16);
  return u.v;
}

// Same fragment but from an LDS tile with LROW-byte row stride.
__device__ __forceinline__ v16h lds_frag(const char* tile, int row0) {
  int lane = threadIdx.x & 31;
  int r = row0 + (lane & 15);
  int hi = lane >> 4;
  const char* p = tile + r * LROW + hi * 16;
  union { v16h v; v8h h[2]; } u;
  u.h[0] = *(const v8h*)(p);
  u.h[1] = *(const v8h*)(p + 32);   // +16 halves
  return u.v;
}

__device__ __forceinline__ float sigmoidf_(float x) {
  return 1.0f / (1.0f + __expf(-x));
}

// ---------- weight conversion: f32 [R,C] -> f16 [C,R] (transpose) ----------
__global__ void transpose_to_f16_kernel(const float* __restrict__ src,
                                        _Float16* __restrict__ dst,
                                        int R, int C) {
  __shared__ float tile[32][33];
  int c0 = blockIdx.x * 32, r0 = blockIdx.y * 32;
  int tx = threadIdx.x, ty = threadIdx.y;   // 32 x 8
#pragma unroll
  for (int i = 0; i < 4; ++i) {
    int r = r0 + ty + i * 8;
    tile[ty + i * 8][tx] = (r < R && c0 + tx < C) ? src[(size_t)r * C + c0 + tx] : 0.0f;
  }
  __syncthreads();
#pragma unroll
  for (int i = 0; i < 4; ++i) {
    int c = c0 + ty + i * 8;
    if (c < C && r0 + tx < R)
      dst[(size_t)c * R + r0 + tx] = (_Float16)tile[tx][ty + i * 8];
  }
}

// head_w [HID, 62] f32 -> head_wT [64, HID] f16 (zero-padded rows 62,63)
__global__ void head_transpose_kernel(const float* __restrict__ head_w,
                                      _Float16* __restrict__ dst) {
  int idx = blockIdx.x * 256 + threadIdx.x;   // 64*1024 elements
  int n = idx >> 10, k = idx & 1023;
  dst[idx] = (n < VOCAB) ? (_Float16)head_w[(size_t)k * VOCAB + n] : (_Float16)0.0f;
}

// ---------- embedding lookup + RMSNorm -> f16 ----------
__global__ void embed_rmsnorm_kernel(const int* __restrict__ tokens,
                                     const float* __restrict__ embed_w,
                                     const float* __restrict__ norm_w,
                                     _Float16* __restrict__ xn) {
  int row = blockIdx.x;
  int tok = tokens[row];
  const float* src = embed_w + (size_t)tok * HID;
  __shared__ float red[256];
  float vals[4];
  float ss = 0.0f;
#pragma unroll
  for (int i = 0; i < 4; ++i) {
    vals[i] = src[threadIdx.x + i * 256];
    ss += vals[i] * vals[i];
  }
  red[threadIdx.x] = ss;
  __syncthreads();
  for (int off = 128; off > 0; off >>= 1) {
    if (threadIdx.x < (unsigned)off) red[threadIdx.x] += red[threadIdx.x + off];
    __syncthreads();
  }
  float scale = rsqrtf(red[0] * (1.0f / HID) + 1e-6f);
#pragma unroll
  for (int i = 0; i < 4; ++i) {
    int c = threadIdx.x + i * 256;
    xn[(size_t)row * HID + c] = (_Float16)(vals[i] * scale * norm_w[c]);
  }
}

// ---------- GEMM1 (gate-fused, async-LDS staged) ----------
// xn[4096,1024] x in_wT[16384,1024]^T -> a, bx, cg [4096,4096] f32.
// Block: 256 thr (8 waves, 4M x 2N). Block tile 128M x 64Ng x 4 groups.
// LDS: double-buffered {A:128 rows, B:256 rows} x 80B.
#define G1_ROWS (128 + 256)
#define G1_BUF  (G1_ROWS * LROW)          // 30720 B
__global__ __launch_bounds__(256)
void gemm1_fused_kernel(const _Float16* __restrict__ XN,
                        const _Float16* __restrict__ WT,   // [PROJ, HID] f16
                        const float* __restrict__ in_b,
                        float* __restrict__ a_buf,
                        float* __restrict__ bx_buf,
                        float* __restrict__ cg_buf) {
  __shared__ __align__(16) char smem[2 * G1_BUF];
  const unsigned lds_base = (unsigned)(uintptr_t)&smem[0];

  const int tid = threadIdx.x;
  const int w = tid >> 5;
  const int wm = w >> 1;                 // 0..3
  const int wn = w & 1;                  // 0..1
  const int mBlock = blockIdx.y * 128;
  const int nBase  = blockIdx.x * 64;
  const int m0 = mBlock + wm * 32;
  const int n0w = wn * 32;               // within 64-wide group tile

  // Stage one k-step: A rows [mBlock..+128), B rows = 4 groups x 64 cols.
  auto stage = [&](int k0, int buf) {
    unsigned aBase = lds_base + buf * G1_BUF;
    unsigned bBase = aBase + 128 * LROW;
#pragma unroll
    for (int u = 0; u < 2; ++u) {              // 512 A segs
      int seg = tid + u * 256;
      int r = seg >> 2, s = seg & 3;
      async_copy16(aBase + r * LROW + s * 16,
                   XN + (size_t)(mBlock + r) * HID + k0 + s * 8);
    }
#pragma unroll
    for (int u = 0; u < 4; ++u) {              // 1024 B segs
      int seg = tid + u * 256;
      int r = seg >> 2, s = seg & 3;
      int g = r >> 6, nw = r & 63;
      async_copy16(bBase + r * LROW + s * 16,
                   WT + (size_t)(g * STATE + nBase + nw) * HID + k0 + s * 8);
    }
  };

  v8f acc[4][2][2] = {};
  stage(0, 0);
  const int nk = HID / 32;
  for (int kstep = 0; kstep < nk; ++kstep) {
    int cur = kstep & 1;
    wait_async0();
    __syncthreads();
    if (kstep + 1 < nk) stage((kstep + 1) * 32, cur ^ 1);

    const char* aTile = smem + cur * G1_BUF;
    const char* bTile = aTile + 128 * LROW;
    v16h a0 = lds_frag(aTile, wm * 32);
    v16h a1 = lds_frag(aTile, wm * 32 + 16);
#pragma unroll
    for (int g = 0; g < 4; ++g) {
#pragma unroll
      for (int j = 0; j < 2; ++j) {
        v16h b = lds_frag(bTile, g * 64 + n0w + j * 16);
        acc[g][0][j] = wmma32(a0, b, acc[g][0][j]);
        acc[g][1][j] = wmma32(a1, b, acc[g][1][j]);
      }
    }
    __syncthreads();
  }

  int lane = tid & 31;
  int col0 = lane & 15;
  int rb = (lane >> 4) * 8;
#pragma unroll
  for (int i = 0; i < 2; ++i) {
#pragma unroll
    for (int j = 0; j < 2; ++j) {
      int col = nBase + n0w + j * 16 + col0;
      float b0 = in_b[col];
      float b1 = in_b[STATE + col];
      float b2 = in_b[2 * STATE + col];
      float b3 = in_b[3 * STATE + col];
#pragma unroll
      for (int e = 0; e < 8; ++e) {
        int row = m0 + i * 16 + rb + e;
        float xg = acc[0][i][j][e] + b0;
        float av = sigmoidf_(acc[1][i][j][e] + b1);
        float bv = sigmoidf_(acc[2][i][j][e] + b2);
        float cv = sigmoidf_(acc[3][i][j][e] + b3);
        size_t idx = (size_t)row * STATE + col;
        a_buf[idx]  = av;
        bx_buf[idx] = bv * xg;
        cg_buf[idx] = cv;
      }
    }
  }
}

// ---------- chunked linear-recurrence scan over L ----------
__global__ void scan_phaseA_kernel(const float* __restrict__ a_buf,
                                   const float* __restrict__ bx_buf,
                                   float* __restrict__ cA,
                                   float* __restrict__ cH) {
  int tid = blockIdx.x * 256 + threadIdx.x;       // 2*32*4096
  int s = tid & (STATE - 1);
  int tmp = tid >> 12;
  int chunk = tmp & (NCHUNK - 1);
  int b = tmp >> 5;
  size_t base = ((size_t)(b * SEQ + chunk * CHUNKT)) * STATE + s;
  float ap = 1.0f, h = 0.0f;
  for (int t = 0; t < CHUNKT; ++t) {
    float a = a_buf[base + (size_t)t * STATE];
    float x = bx_buf[base + (size_t)t * STATE];
    h = a * h + x;
    ap *= a;
  }
  size_t ci = ((size_t)(b * NCHUNK + chunk)) * STATE + s;
  cA[ci] = ap;
  cH[ci] = h;
}

__global__ void scan_phaseB_kernel(const float* __restrict__ cA,
                                   const float* __restrict__ cH,
                                   float* __restrict__ pre) {
  int tid = blockIdx.x * 256 + threadIdx.x;       // 2*4096
  int s = tid & (STATE - 1);
  int b = tid >> 12;
  float h = 0.0f;
  for (int c = 0; c < NCHUNK; ++c) {
    size_t ci = ((size_t)(b * NCHUNK + c)) * STATE + s;
    pre[ci] = h;
    h = cA[ci] * h + cH[ci];
  }
}

__global__ void scan_phaseC_kernel(const float* __restrict__ a_buf,
                                   const float* __restrict__ bx_buf,
                                   const float* __restrict__ cg_buf,
                                   const float* __restrict__ pre,
                                   _Float16* __restrict__ y) {
  int tid = blockIdx.x * 256 + threadIdx.x;
  int s = tid & (STATE - 1);
  int tmp = tid >> 12;
  int chunk = tmp & (NCHUNK - 1);
  int b = tmp >> 5;
  size_t base = ((size_t)(b * SEQ + chunk * CHUNKT)) * STATE + s;
  size_t ci = ((size_t)(b * NCHUNK + chunk)) * STATE + s;
  float h = pre[ci];
  for (int t = 0; t < CHUNKT; ++t) {
    size_t idx = base + (size_t)t * STATE;
    float a = a_buf[idx];
    float x = bx_buf[idx];
    h = a * h + x;
    y[idx] = (_Float16)(cg_buf[idx] * h);
  }
}

// ---------- GEMM2 (async-LDS staged): y[4096,4096] x out_wT -> f16 [4096,1024] ----------
// Block 256 thr (8 waves, 4M x 2N); block tile 128M x 128N.
// LDS: double-buffered {A:128 rows, B:128 rows} x 80B.
#define G2_ROWS (128 + 128)
#define G2_BUF  (G2_ROWS * LROW)          // 20480 B
__global__ __launch_bounds__(256)
void gemm2_kernel(const _Float16* __restrict__ Y,
                  const _Float16* __restrict__ WT,     // [HID, STATE] f16
                  const float* __restrict__ out_b,
                  const int* __restrict__ tokens,
                  const float* __restrict__ embed_w,
                  _Float16* __restrict__ OutH) {
  __shared__ __align__(16) char smem[2 * G2_BUF];
  const unsigned lds_base = (unsigned)(uintptr_t)&smem[0];

  const int tid = threadIdx.x;
  const int w = tid >> 5;
  const int wm = w >> 1;                // 0..3
  const int wn = w & 1;                 // 0..1
  const int mBlock = blockIdx.y * 128;
  const int nBlock = blockIdx.x * 128;
  const int m0 = mBlock + wm * 32;

  auto stage = [&](int k0, int buf) {
    unsigned aBase = lds_base + buf * G2_BUF;
    unsigned bBase = aBase + 128 * LROW;
#pragma unroll
    for (int u = 0; u < 2; ++u) {             // 512 A segs
      int seg = tid + u * 256;
      int r = seg >> 2, s = seg & 3;
      async_copy16(aBase + r * LROW + s * 16,
                   Y + (size_t)(mBlock + r) * STATE + k0 + s * 8);
    }
#pragma unroll
    for (int u = 0; u < 2; ++u) {             // 512 B segs
      int seg = tid + u * 256;
      int r = seg >> 2, s = seg & 3;
      async_copy16(bBase + r * LROW + s * 16,
                   WT + (size_t)(nBlock + r) * STATE + k0 + s * 8);
    }
  };

  v8f acc[2][4] = {};
  stage(0, 0);
  const int nk = STATE / 32;
  for (int kstep = 0; kstep < nk; ++kstep) {
    int cur = kstep & 1;
    wait_async0();
    __syncthreads();
    if (kstep + 1 < nk) stage((kstep + 1) * 32, cur ^ 1);

    const char* aTile = smem + cur * G2_BUF;
    const char* bTile = aTile + 128 * LROW;
    v16h a0 = lds_frag(aTile, wm * 32);
    v16h a1 = lds_frag(aTile, wm * 32 + 16);
#pragma unroll
    for (int j = 0; j < 4; ++j) {
      v16h b = lds_frag(bTile, wn * 64 + j * 16);
      acc[0][j] = wmma32(a0, b, acc[0][j]);
      acc[1][j] = wmma32(a1, b, acc[1][j]);
    }
    __syncthreads();
  }

  int lane = tid & 31;
  int col0 = lane & 15;
  int rb = (lane >> 4) * 8;
#pragma unroll
  for (int i = 0; i < 2; ++i) {
#pragma unroll
    for (int j = 0; j < 4; ++j) {
      int col = nBlock + wn * 64 + j * 16 + col0;
      float bias = out_b[col];
#pragma unroll
      for (int e = 0; e < 8; ++e) {
        int row = m0 + i * 16 + rb + e;
        float res = embed_w[(size_t)tokens[row] * HID + col];
        OutH[(size_t)row * HID + col] = (_Float16)(acc[i][j][e] + bias + res);
      }
    }
  }
}

// ---------- GEMM3: out_h[4096,1024] x head_wT[64,1024] + head_b -> logits [4096,62] ----------
__global__ __launch_bounds__(128)
void gemm3_kernel(const _Float16* __restrict__ A,
                  const _Float16* __restrict__ WT,   // [64, HID]
                  const float* __restrict__ head_b,
                  float* __restrict__ out) {
  int w = threadIdx.x >> 5;
  int m0 = blockIdx.x * 128 + w * 32;
  v8f acc[2][4] = {};
  for (int k0 = 0; k0 < HID; k0 += 32) {
    v16h a0 = load_frag(A + (size_t)m0 * HID + k0, HID);
    v16h a1 = load_frag(A + (size_t)(m0 + 16) * HID + k0, HID);
#pragma unroll
    for (int j = 0; j < 4; ++j) {
      v16h b = load_frag(WT + (size_t)(j * 16) * HID + k0, HID);
      acc[0][j] = wmma32(a0, b, acc[0][j]);
      acc[1][j] = wmma32(a1, b, acc[1][j]);
    }
  }
  int lane = threadIdx.x & 31;
  int col0 = lane & 15;
  int rb = (lane >> 4) * 8;
#pragma unroll
  for (int i = 0; i < 2; ++i) {
#pragma unroll
    for (int j = 0; j < 4; ++j) {
      int col = j * 16 + col0;
      if (col < VOCAB) {
        float hb = head_b[col];
#pragma unroll
        for (int e = 0; e < 8; ++e) {
          int row = m0 + i * 16 + rb + e;
          out[(size_t)row * VOCAB + col] = acc[i][j][e] + hb;
        }
      }
    }
  }
}

extern "C" void kernel_launch(void* const* d_in, const int* in_sizes, int n_in,
                              void* d_out, int out_size, void* d_ws, size_t ws_size,
                              hipStream_t stream) {
  const int*   tokens  = (const int*)d_in[0];
  const float* embed_w = (const float*)d_in[1];
  const float* norm_w  = (const float*)d_in[2];
  const float* in_w    = (const float*)d_in[3];
  const float* in_b    = (const float*)d_in[4];
  const float* out_w   = (const float*)d_in[5];
  const float* out_b   = (const float*)d_in[6];
  const float* head_w  = (const float*)d_in[7];
  const float* head_b  = (const float*)d_in[8];
  float* out = (float*)d_out;

  char* ws = (char*)d_ws;
  size_t off = 0;
  auto alloc = [&](size_t bytes) -> void* {
    void* p = ws + off;
    off += (bytes + 255) & ~(size_t)255;
    return p;
  };
  _Float16* xn      = (_Float16*)alloc((size_t)ROWS * HID * 2);       //   8 MiB
  _Float16* in_wT   = (_Float16*)alloc((size_t)PROJ * HID * 2);       //  32 MiB
  _Float16* out_wT  = (_Float16*)alloc((size_t)HID * STATE * 2);      //   8 MiB
  _Float16* head_wT = (_Float16*)alloc((size_t)64 * HID * 2);         // 128 KiB
  float*    a_buf   = (float*)alloc((size_t)ROWS * STATE * 4);        //  64 MiB
  float*    bx_buf  = (float*)alloc((size_t)ROWS * STATE * 4);        //  64 MiB
  float*    cg_buf  = (float*)alloc((size_t)ROWS * STATE * 4);        //  64 MiB
  float*    cA      = (float*)alloc((size_t)BATCH * NCHUNK * STATE * 4);
  float*    cH      = (float*)alloc((size_t)BATCH * NCHUNK * STATE * 4);
  float*    pre     = (float*)alloc((size_t)BATCH * NCHUNK * STATE * 4);
  _Float16* y_h     = (_Float16*)alloc((size_t)ROWS * STATE * 2);     //  32 MiB
  _Float16* out_h   = (_Float16*)alloc((size_t)ROWS * HID * 2);       //   8 MiB
  (void)ws_size; (void)in_sizes; (void)n_in; (void)out_size;

  dim3 tb(32, 8);
  transpose_to_f16_kernel<<<dim3(PROJ / 32, HID / 32), tb, 0, stream>>>(in_w, in_wT, HID, PROJ);
  transpose_to_f16_kernel<<<dim3(HID / 32, STATE / 32), tb, 0, stream>>>(out_w, out_wT, STATE, HID);
  head_transpose_kernel<<<(64 * HID) / 256, 256, 0, stream>>>(head_w, head_wT);

  embed_rmsnorm_kernel<<<ROWS, 256, 0, stream>>>(tokens, embed_w, norm_w, xn);

  gemm1_fused_kernel<<<dim3(STATE / 64, ROWS / 128), 256, 0, stream>>>(
      xn, in_wT, in_b, a_buf, bx_buf, cg_buf);

  scan_phaseA_kernel<<<(BATCH * NCHUNK * STATE) / 256, 256, 0, stream>>>(a_buf, bx_buf, cA, cH);
  scan_phaseB_kernel<<<(BATCH * STATE) / 256, 256, 0, stream>>>(cA, cH, pre);
  scan_phaseC_kernel<<<(BATCH * NCHUNK * STATE) / 256, 256, 0, stream>>>(
      a_buf, bx_buf, cg_buf, pre, y_h);

  gemm2_kernel<<<dim3(HID / 128, ROWS / 128), 256, 0, stream>>>(
      y_h, out_wT, out_b, tokens, embed_w, out_h);

  gemm3_kernel<<<ROWS / 128, 128, 0, stream>>>(out_h, head_wT, head_b, out);
}
